// LorentzSelfAttentionBlock_51419348468124
// MI455X (gfx1250) — matrix operand order repaired and checked by reading, not compiled, not run on hardware
//
#include <hip/hip_runtime.h>
#include <hip/hip_bf16.h>
#include <stdint.h>

// ---------------------------------------------------------------------------
// Types for CDNA5 WMMA (wave32, v_wmma_f32_16x16x32_bf16)
// ---------------------------------------------------------------------------
typedef __bf16 bf16_t;
typedef bf16_t v16bf __attribute__((ext_vector_type(16)));
typedef bf16_t v8bf  __attribute__((ext_vector_type(8)));
typedef float  v8f   __attribute__((ext_vector_type(8)));

#define EPS_F 1e-6f

__device__ __forceinline__ bf16_t f2bf(float f) {
    __hip_bfloat16 h = __float2bfloat16(f);
    return *reinterpret_cast<bf16_t*>(&h);
}

__device__ __forceinline__ float wave_sum(float v) {
#pragma unroll
    for (int o = 16; o > 0; o >>= 1) v += __shfl_xor(v, o, 32);
    return v;
}
__device__ __forceinline__ float wave_max(float v) {
#pragma unroll
    for (int o = 16; o > 0; o >>= 1) v = fmaxf(v, __shfl_xor(v, o, 32));
    return v;
}
__device__ __forceinline__ float block_sum(float v, float* sbuf) {
    int lane = threadIdx.x & 31, w = threadIdx.x >> 5;
    int nw = (int)(blockDim.x >> 5);
    v = wave_sum(v);
    if (lane == 0) sbuf[w] = v;
    __syncthreads();
    float r = ((int)threadIdx.x < nw) ? sbuf[threadIdx.x] : 0.0f;
    if (w == 0) { r = wave_sum(r); if (lane == 0) sbuf[0] = r; }
    __syncthreads();
    float out = sbuf[0];
    __syncthreads();
    return out;
}
__device__ __forceinline__ float block_max(float v, float* sbuf) {
    int lane = threadIdx.x & 31, w = threadIdx.x >> 5;
    int nw = (int)(blockDim.x >> 5);
    v = wave_max(v);
    if (lane == 0) sbuf[w] = v;
    __syncthreads();
    float r = ((int)threadIdx.x < nw) ? sbuf[threadIdx.x] : -3.4e38f;
    if (w == 0) { r = wave_max(r); if (lane == 0) sbuf[0] = r; }
    __syncthreads();
    float out = sbuf[0];
    __syncthreads();
    return out;
}

// CDNA5 async global -> LDS copy (ASYNCcnt path), 16B per lane.
__device__ __forceinline__ void async_load_b128(uint32_t lds_off, uint64_t gaddr) {
    asm volatile("global_load_async_to_lds_b128 %0, %1, off"
                 :: "v"(lds_off), "v"(gaddr) : "memory");
}
__device__ __forceinline__ void wait_async0() {
    asm volatile("s_wait_asynccnt 0x0" ::: "memory");
}

// ---------------------------------------------------------------------------
// LDS-staged, double-buffered bf16 WMMA GEMM:
//   C[M x N](f32) = A[M x K](bf16, row-major lda) @ Bt[N x K](bf16)^T
// Block = 256 thr = 8 waves; block tile 128(M) x 128(N); stage K = 64.
// LDS per stage: A(128x64) + B(128x64), each row padded 64->72 elems
// (144B stride => all 16 fragment lanes hit distinct banks). 2 stages = 72KB.
// Per stage each thread issues 8 global_load_async_to_lds_b128 (16B/lane),
// prefetching stage s+1 while computing stage s (2 K-steps, 16 WMMAs) from
// LDS; one s_wait_asynccnt 0 + barrier per stage closes the pipeline.
// Each wave: 64x32 sub-tile = 8 accumulators.
// Requires M%128==0, N%128==0, K%64==0 (guaranteed by padded layouts).
// ---------------------------------------------------------------------------
#define LROW   72          // padded LDS row, elements
#define LTILE  9216        // 128*72 elements per tile
#define LSTAGE 18432       // elements per stage (A+B)

__global__ __launch_bounds__(256) void gemm_bf16_kernel(
    const bf16_t* __restrict__ A, long lda, long sAz,
    const bf16_t* __restrict__ Bt, long ldb, long sBz,
    float* __restrict__ C, long ldc, long sCz,
    int K)
{
    __shared__ __align__(16) bf16_t smem[2 * LSTAGE]; // 73728 B

    const bf16_t* Ab = A + (long)blockIdx.z * sAz;
    const bf16_t* Bb = Bt + (long)blockIdx.z * sBz;
    float* Cb = C + (long)blockIdx.z * sCz;

    const long m_blk = (long)blockIdx.y * 128;
    const long n_blk = (long)blockIdx.x * 128;

    const int t = threadIdx.x;
    const int wave = t >> 5;
    const int lane = t & 31;
    const int wm = wave & 1;        // 2 waves along M
    const int wn = wave >> 1;       // 4 waves along N
    const int m0w = wm * 64;
    const int n0w = wn * 32;
    const int half = lane >> 4;     // 0: lanes 0-15, 1: lanes 16-31
    const int l16 = lane & 15;

    // --- staging: per stage 1024 A-chunks + 1024 B-chunks of 16B ---
    // thread t handles chunk ids {t, t+256, t+512, t+768}: row = id>>3, sub = id&7
    const int row0 = t >> 3;            // 0..31
    const int sub  = t & 7;             // 16B chunk within 128B row slice
    const uint32_t lds_base = (uint32_t)(uintptr_t)(&smem[0]);
    uint32_t ldsA[4], ldsB[4];
    uint64_t gA[4], gB[4];
#pragma unroll
    for (int i = 0; i < 4; i++) {
        int row = row0 + 32 * i;
        ldsA[i] = lds_base + (uint32_t)(row * 144 + sub * 16);
        ldsB[i] = lds_base + 36864u + (uint32_t)(row * 144 + sub * 16);
        gA[i] = (uint64_t)(uintptr_t)(Ab + (m_blk + row) * lda + sub * 8);
        gB[i] = (uint64_t)(uintptr_t)(Bb + (n_blk + row) * ldb + sub * 8);
    }

    v8f acc[4][2] = {};

    // prologue: stage 0
    {
#pragma unroll
        for (int i = 0; i < 4; i++) { async_load_b128(ldsA[i], gA[i]); async_load_b128(ldsB[i], gB[i]); }
        wait_async0();
        __syncthreads();
    }

    const int ns = K >> 6;              // stages of 64
    for (int s = 0; s < ns; s++) {
        const int cs = s & 1;
        // prefetch next stage into the other buffer (safe: end-of-prev-iter
        // barrier guarantees that buffer's readers are done)
        if (s + 1 < ns) {
            const uint32_t so = (uint32_t)((s + 1) & 1) * 36864u;  // stage bytes
            const uint64_t go = (uint64_t)(s + 1) * 128u;          // 64 elems * 2B
#pragma unroll
            for (int i = 0; i < 4; i++) {
                async_load_b128(ldsA[i] + so, gA[i] + go);
                async_load_b128(ldsB[i] + so, gB[i] + go);
            }
        }

        // compute current stage from LDS: 2 K-steps of 32
        const bf16_t* sA = &smem[cs * LSTAGE];
        const bf16_t* sB = &smem[cs * LSTAGE + LTILE];
#pragma unroll
        for (int ks = 0; ks < 2; ks++) {
            v16bf af[4], bfr[2];
#pragma unroll
            for (int i = 0; i < 4; i++) {
                const bf16_t* p = sA + (m0w + 16 * i + l16) * LROW + ks * 32 + half * 8;
                v8bf lo = *(const v8bf*)(p);
                v8bf hi = *(const v8bf*)(p + 16);
                af[i] = __builtin_shufflevector(lo, hi, 0,1,2,3,4,5,6,7,8,9,10,11,12,13,14,15);
            }
#pragma unroll
            for (int j = 0; j < 2; j++) {
                const bf16_t* p = sB + (n0w + 16 * j + l16) * LROW + ks * 32 + half * 8;
                v8bf lo = *(const v8bf*)(p);
                v8bf hi = *(const v8bf*)(p + 16);
                bfr[j] = __builtin_shufflevector(lo, hi, 0,1,2,3,4,5,6,7,8,9,10,11,12,13,14,15);
            }
#pragma unroll
            for (int i = 0; i < 4; i++)
#pragma unroll
                for (int j = 0; j < 2; j++)
                    acc[i][j] = __builtin_amdgcn_wmma_f32_16x16x32_bf16(
                        false, af[i], false, bfr[j], (short)0, acc[i][j], false, false);
        }

        wait_async0();
        __syncthreads();
    }

    // C store (f32 16x16 layout: VGPR r -> M=r / r+8 by lane half, N=lane&15)
#pragma unroll
    for (int i = 0; i < 4; i++) {
        long r0 = m_blk + m0w + 16 * i + half * 8;
#pragma unroll
        for (int j = 0; j < 2; j++) {
            long c0 = n_blk + n0w + 16 * j + l16;
#pragma unroll
            for (int r = 0; r < 8; r++)
                Cb[(r0 + r) * ldc + c0] = acc[i][j][r];
        }
    }
}

// ---------------------------------------------------------------------------
// Weight repacks (f32 -> bf16, transposed to N x Kpad, zero-padded K)
// ---------------------------------------------------------------------------
__global__ void repack_qkv_kernel(const float* __restrict__ Wq,
                                  const float* __restrict__ Wk,
                                  const float* __restrict__ Wv,
                                  bf16_t* __restrict__ out /* 3072 x 1088 */)
{
    int n = blockIdx.x;            // 0..3071
    int which = n >> 10;           // 0=q 1=k 2=v
    int h = (n & 1023) >> 6;
    int e = n & 63;
    const float* W = (which == 0) ? Wq : ((which == 1) ? Wk : Wv);
    const float* src = W + (long)h * 1025 * 64 + e;   // W[h][d][e]
    bf16_t* dst = out + (long)n * 1088;
    for (int d = threadIdx.x; d < 1088; d += blockDim.x)
        dst[d] = f2bf(d < 1025 ? src[(long)d * 64] : 0.0f);
}

__global__ void repack_t_kernel(const float* __restrict__ W /* K0 x N */,
                                int K0, int N, int Kpad,
                                bf16_t* __restrict__ out /* N x Kpad */)
{
    int n = blockIdx.x;
    bf16_t* dst = out + (long)n * Kpad;
    for (int d = threadIdx.x; d < Kpad; d += blockDim.x)
        dst[d] = f2bf(d < K0 ? W[(long)d * N + n] : 0.0f);
}

// ---------------------------------------------------------------------------
// Hyperbolic layernorm: x(T x 1025) -> xn(T x 1088) bf16 = [t, s_norm, 0pad]
// ---------------------------------------------------------------------------
__global__ __launch_bounds__(256) void hyp_ln_kernel(
    const float* __restrict__ x, const float* __restrict__ g,
    const float* __restrict__ b, bf16_t* __restrict__ xn)
{
    __shared__ float sbuf[8];
    long tok = blockIdx.x;
    const float* row = x + tok * 1025 + 1;   // space part
    float v[4]; float s = 0;
#pragma unroll
    for (int i = 0; i < 4; i++) { v[i] = row[threadIdx.x + 256 * i]; s += v[i]; }
    float mu = block_sum(s, sbuf) * (1.0f / 1024.0f);
    float ss = 0;
#pragma unroll
    for (int i = 0; i < 4; i++) { float d = v[i] - mu; ss += d * d; }
    float var = block_sum(ss, sbuf) * (1.0f / 1024.0f);
    float rs = rsqrtf(var + 1e-5f);
    float sn[4]; float q = 0;
#pragma unroll
    for (int i = 0; i < 4; i++) {
        int c = threadIdx.x + 256 * i;
        sn[i] = (v[i] - mu) * rs * g[c] + b[c];
        q += sn[i] * sn[i];
    }
    float qs = block_sum(q, sbuf);
    bf16_t* out = xn + tok * 1088;
    if (threadIdx.x == 0) out[0] = f2bf(sqrtf(qs + 1.0f));
#pragma unroll
    for (int i = 0; i < 4; i++) out[1 + threadIdx.x + 256 * i] = f2bf(sn[i]);
    if (threadIdx.x < 63) out[1025 + threadIdx.x] = f2bf(0.0f);
}

// ---------------------------------------------------------------------------
// QKV epilogue: bias + RoPE + normalize + lift; writes
//   qb[b][h][s][128] = [-t_q, qhat(64), 0pad]  (Lorentz sign baked into q)
//   kb[b][h][s][128] = [ t_k, khat(64), 0pad]
//   vt[b][h][128][S] transposed V' = [t_v; v(64); 0pad rows]
// ---------------------------------------------------------------------------
__global__ __launch_bounds__(64) void qkv_pack_kernel(
    const float* __restrict__ qkv, // T x 3072 (q|k|v each h*64+e)
    const float* __restrict__ bq, const float* __restrict__ bk,
    const float* __restrict__ bv,
    const float* __restrict__ rope_cos, const float* __restrict__ rope_sin, // S x 32
    bf16_t* __restrict__ qb, bf16_t* __restrict__ kb, bf16_t* __restrict__ vt,
    int S, int H)
{
    __shared__ float sbuf[8];
    int s = blockIdx.x, h = blockIdx.y, b = blockIdx.z;
    int e = threadIdx.x;
    long tok = (long)b * S + s;
    const float* base = qkv + tok * 3072;
    int he = h * 64 + e;
    float q = base[he] + bq[he];
    float k = base[1024 + he] + bk[he];
    float v = base[2048 + he] + bv[he];
    float c  = rope_cos[(long)s * 32 + (e >> 1)];
    float sn = rope_sin[(long)s * 32 + (e >> 1)];
    float qp = __shfl_xor(q, 1, 32);
    float kp = __shfl_xor(k, 1, 32);
    float qr = ((e & 1) == 0) ? (q * c - qp * sn) : (qp * sn + q * c);
    float kr = ((e & 1) == 0) ? (k * c - kp * sn) : (kp * sn + k * c);
    float qss = block_sum(qr * qr, sbuf);
    float kss = block_sum(kr * kr, sbuf);
    float vss = block_sum(v * v, sbuf);
    float qinv = rsqrtf(qss + EPS_F);
    float kinv = rsqrtf(kss + EPS_F);
    float tq = sqrtf(1.0f + qss * qinv * qinv);
    float tk = sqrtf(1.0f + kss * kinv * kinv);
    float tv = sqrtf(1.0f + vss);
    long obase = (((long)b * H + h) * S + s) * 128;
    qb[obase + 1 + e] = f2bf(qr * qinv);
    kb[obase + 1 + e] = f2bf(kr * kinv);
    if (e == 0) { qb[obase] = f2bf(-tq); kb[obase] = f2bf(tk); }
    if (e >= 1) { qb[obase + 64 + e] = f2bf(0.0f); kb[obase + 64 + e] = f2bf(0.0f); } // 65..127
    long vbase = ((long)b * H + h) * 128 * (long)S + s;
    vt[vbase + (long)(1 + e) * S] = f2bf(v);
    if (e == 0) vt[vbase] = f2bf(tv);
    if (e < 63) vt[vbase + (long)(65 + e) * S] = f2bf(0.0f);   // pad rows 65..127
}

// ---------------------------------------------------------------------------
// Softmax over keys: att = (2 + 2*cin)/scale + bias; write P in bf16
// ---------------------------------------------------------------------------
__global__ __launch_bounds__(256) void softmax_kernel(
    const float* __restrict__ scores, bf16_t* __restrict__ P,
    const float* __restrict__ scale_p, const float* __restrict__ bias_p, int S)
{
    __shared__ float sbuf[8];
    int srow = blockIdx.x, h = blockIdx.y;
    const float* row = scores + ((long)h * S + srow) * S;
    bf16_t* prow = P + ((long)h * S + srow) * S;
    float inv_scale = 1.0f / scale_p[0];
    float bias = bias_p[0];
    float a[4]; float mx = -3.4e38f;
#pragma unroll
    for (int i = 0; i < 4; i++) {
        float cin = row[threadIdx.x + 256 * i];
        a[i] = (2.0f + 2.0f * cin) * inv_scale + bias;
        mx = fmaxf(mx, a[i]);
    }
    mx = block_max(mx, sbuf);
    float ssum = 0;
#pragma unroll
    for (int i = 0; i < 4; i++) { a[i] = __expf(a[i] - mx); ssum += a[i]; }
    float inv = 1.0f / block_sum(ssum, sbuf);
#pragma unroll
    for (int i = 0; i < 4; i++) prow[threadIdx.x + 256 * i] = f2bf(a[i] * inv);
}

// ---------------------------------------------------------------------------
// Per-head project(m) + concat into cat (T x 1088 bf16, cols h*65+j)
// ---------------------------------------------------------------------------
__global__ __launch_bounds__(128) void cat_project_kernel(
    const float* __restrict__ m /* [h][s][128] */, bf16_t* __restrict__ cat,
    int S, int b)
{
    __shared__ float sbuf[8];
    int s = blockIdx.x, h = blockIdx.y;
    const float* row = m + ((long)h * S + s) * 128;
    int j = threadIdx.x;
    float val = (j < 65) ? row[j] : 0.0f;
    float signed_sq = (j == 0) ? -(val * val) : (val * val); // sums to l_inner
    float li = block_sum((j < 65) ? signed_sq : 0.0f, sbuf);
    float invd = rsqrtf(fmaxf(-li, EPS_F));
    long tok = (long)b * S + s;
    if (j < 65) cat[tok * 1088 + h * 65 + j] = f2bf(val * invd);
    if (h == 0 && j >= 65 && j < 113) cat[tok * 1088 + 1040 + (j - 65)] = f2bf(0.0f); // 1040..1087
}

// ---------------------------------------------------------------------------
// Residual + to_manifold + project:  xout = project(xin + w*[t, sp+bias])
// ---------------------------------------------------------------------------
__global__ __launch_bounds__(256) void residual_project_kernel(
    const float* __restrict__ xin,  // T x 1025
    const float* __restrict__ sp,   // T x 1024 (GEMM out, pre-bias space part)
    const float* __restrict__ bias, // 1024
    const float* __restrict__ wres_p,
    float* __restrict__ xout)       // T x 1025
{
    __shared__ float sbuf[8];
    long tok = blockIdx.x;
    const float* srow = sp + tok * 1024;
    const float* xrow = xin + tok * 1025;
    float* orow = xout + tok * 1025;
    float wres = wres_p[0];
    float a[4]; float ss = 0;
#pragma unroll
    for (int i = 0; i < 4; i++) {
        int c = threadIdx.x + 256 * i;
        a[i] = srow[c] + bias[c];
        ss += a[i] * a[i];
    }
    float t = sqrtf(1.0f + block_sum(ss, sbuf));
    float z0 = xrow[0] + wres * t;
    float z[4]; float zs = 0;
#pragma unroll
    for (int i = 0; i < 4; i++) {
        z[i] = xrow[1 + threadIdx.x + 256 * i] + wres * a[i];
        zs += z[i] * z[i];
    }
    float zss = block_sum(zs, sbuf);
    float invd = rsqrtf(fmaxf(z0 * z0 - zss, EPS_F));
    if (threadIdx.x == 0) orow[0] = z0 * invd;
#pragma unroll
    for (int i = 0; i < 4; i++)
        orow[1 + threadIdx.x + 256 * i] = z[i] * invd;
}

// ---------------------------------------------------------------------------
// FF epilogue: h = to_manifold(gelu(ff + b1)) -> bf16 (T x 4160)
// ---------------------------------------------------------------------------
__global__ __launch_bounds__(256) void gelu_manifold_kernel(
    const float* __restrict__ ff, const float* __restrict__ b1,
    bf16_t* __restrict__ hbuf)
{
    __shared__ float sbuf[8];
    long tok = blockIdx.x;
    const float* row = ff + tok * 4096;
    float g[16]; float ss = 0;
#pragma unroll
    for (int i = 0; i < 16; i++) {
        int c = threadIdx.x + 256 * i;
        float v = row[c] + b1[c];
        float u = 0.7978845608028654f * (v + 0.044715f * v * v * v);
        float gg = 0.5f * v * (1.0f + tanhf(u));
        g[i] = gg; ss += gg * gg;
    }
    float t = sqrtf(1.0f + block_sum(ss, sbuf));
    bf16_t* out = hbuf + tok * 4160;
    if (threadIdx.x == 0) out[0] = f2bf(t);
#pragma unroll
    for (int i = 0; i < 16; i++) out[1 + threadIdx.x + 256 * i] = f2bf(g[i]);
    if (threadIdx.x < 63) out[4097 + threadIdx.x] = f2bf(0.0f);  // 4097..4159
}

// ---------------------------------------------------------------------------
// Host orchestration
// ---------------------------------------------------------------------------
extern "C" void kernel_launch(void* const* d_in, const int* in_sizes, int n_in,
                              void* d_out, int out_size, void* d_ws, size_t ws_size,
                              hipStream_t stream)
{
    (void)in_sizes; (void)n_in; (void)out_size; (void)ws_size;
    const float* x        = (const float*)d_in[0];
    const float* rope_cos = (const float*)d_in[1];
    const float* rope_sin = (const float*)d_in[2];
    const float* norm1_g  = (const float*)d_in[3];
    const float* norm1_b  = (const float*)d_in[4];
    const float* Wq       = (const float*)d_in[5];
    const float* bq       = (const float*)d_in[6];
    const float* Wk       = (const float*)d_in[7];
    const float* bk       = (const float*)d_in[8];
    const float* Wv       = (const float*)d_in[9];
    const float* bv       = (const float*)d_in[10];
    const float* attn_scale = (const float*)d_in[11];
    const float* attn_bias  = (const float*)d_in[12];
    const float* Wo       = (const float*)d_in[13];
    const float* bo       = (const float*)d_in[14];
    const float* w_res1   = (const float*)d_in[15];
    const float* norm2_g  = (const float*)d_in[16];
    const float* norm2_b  = (const float*)d_in[17];
    const float* W1       = (const float*)d_in[18];
    const float* b1       = (const float*)d_in[19];
    const float* W2       = (const float*)d_in[20];
    const float* b2       = (const float*)d_in[21];
    const float* w_res2   = (const float*)d_in[22];
    float* out = (float*)d_out;

    const int B = 4, S = 1024, H = 16, FF = 4096;
    const int T = B * S;          // 4096 tokens
    const int K1 = 1088;          // 1025 padded to 64-multiple
    const int KQ = 128;           // 65 padded to 64-multiple
    const int K2 = 4160;          // 4097 padded to 64-multiple

    char* cur = (char*)d_ws;
    auto alloc = [&](size_t bytes) -> char* {
        char* p = cur;
        cur += (bytes + 255) & ~(size_t)255;
        return p;
    };
    bf16_t* xn   = (bf16_t*)alloc((size_t)T * K1 * 2);
    bf16_t* wqkv = (bf16_t*)alloc((size_t)3072 * K1 * 2);
    bf16_t* wo_t = (bf16_t*)alloc((size_t)1024 * K1 * 2);
    bf16_t* w1_t = (bf16_t*)alloc((size_t)FF * K1 * 2);
    bf16_t* w2_t = (bf16_t*)alloc((size_t)1024 * K2 * 2);
    bf16_t* qb   = (bf16_t*)alloc((size_t)B * H * S * KQ * 2);
    bf16_t* kb   = (bf16_t*)alloc((size_t)B * H * S * KQ * 2);
    bf16_t* vt   = (bf16_t*)alloc((size_t)B * H * 128 * S * 2);
    bf16_t* catb = (bf16_t*)alloc((size_t)T * K1 * 2);
    bf16_t* hbuf = (bf16_t*)alloc((size_t)T * K2 * 2);
    bf16_t* pbuf = (bf16_t*)alloc((size_t)H * S * S * 2);
    float* big    = (float*)alloc((size_t)T * FF * 4);      // qkv / wo / ff / mlp (reused)
    float* scores = (float*)alloc((size_t)H * S * S * 4);   // per-b reuse
    float* mbuf   = (float*)alloc((size_t)H * S * 128 * 4); // per-b reuse
    float* x1     = (float*)alloc((size_t)T * 1025 * 4);

    // --- Weight repacks (bf16, transposed, K-padded) ---
    repack_qkv_kernel<<<3072, 256, 0, stream>>>(Wq, Wk, Wv, wqkv);
    repack_t_kernel<<<1024, 256, 0, stream>>>(Wo, 1040, 1024, K1, wo_t);
    repack_t_kernel<<<FF,   256, 0, stream>>>(W1, 1025, FF,   K1, w1_t);
    repack_t_kernel<<<1024, 256, 0, stream>>>(W2, 4097, 1024, K2, w2_t);

    // --- LN1 + fused QKV GEMM (4096 x 3072 x 1088) ---
    hyp_ln_kernel<<<T, 256, 0, stream>>>(x, norm1_g, norm1_b, xn);
    gemm_bf16_kernel<<<dim3(3072 / 128, T / 128, 1), 256, 0, stream>>>(
        xn, K1, 0, wqkv, K1, 0, big, 3072, 0, K1);

    // --- RoPE + normalize + manifold pack ---
    qkv_pack_kernel<<<dim3(S, H, B), 64, 0, stream>>>(
        big, bq, bk, bv, rope_cos, rope_sin, qb, kb, vt, S, H);

    // --- Attention per batch (stream-ordered buffer reuse) ---
    const long qks = (long)H * S * KQ;
    const long vts = (long)H * 128 * S;
    for (int b = 0; b < B; b++) {
        // scores[h] = Qb @ Kb^T  (exact Lorentz inner product, K=128 padded)
        gemm_bf16_kernel<<<dim3(S / 128, S / 128, H), 256, 0, stream>>>(
            qb + (long)b * qks, KQ, (long)S * KQ,
            kb + (long)b * qks, KQ, (long)S * KQ,
            scores, S, (long)S * S, KQ);
        softmax_kernel<<<dim3(S, H), 256, 0, stream>>>(scores, pbuf, attn_scale, attn_bias, S);
        // m[h] = P @ V'  (N padded to 128, V' transposed+zero-padded)
        gemm_bf16_kernel<<<dim3(1, S / 128, H), 256, 0, stream>>>(
            pbuf, S, (long)S * S,
            vt + (long)b * vts, S, (long)128 * S,
            mbuf, 128, (long)S * 128, S);
        cat_project_kernel<<<dim3(S, H), 128, 0, stream>>>(mbuf, catb, S, b);
    }

    // --- Wo GEMM + residual/project ---
    gemm_bf16_kernel<<<dim3(1024 / 128, T / 128, 1), 256, 0, stream>>>(
        catb, K1, 0, wo_t, K1, 0, big, 1024, 0, K1);
    residual_project_kernel<<<T, 256, 0, stream>>>(x, big, bo, w_res1, x1);

    // --- LN2 + FF ---
    hyp_ln_kernel<<<T, 256, 0, stream>>>(x1, norm2_g, norm2_b, xn);
    gemm_bf16_kernel<<<dim3(FF / 128, T / 128, 1), 256, 0, stream>>>(
        xn, K1, 0, w1_t, K1, 0, big, FF, 0, K1);
    gelu_manifold_kernel<<<T, 256, 0, stream>>>(big, b1, hbuf);
    gemm_bf16_kernel<<<dim3(1024 / 128, T / 128, 1), 256, 0, stream>>>(
        hbuf, K2, 0, w2_t, K2, 0, big, 1024, 0, K2);
    residual_project_kernel<<<T, 256, 0, stream>>>(x1, big, b2, w_res2, out);
}
